// CrossChannelAttention_30803505447431
// MI455X (gfx1250) — matrix operand order repaired and checked by reading, not compiled, not run on hardware
//
#include <hip/hip_runtime.h>
#include <hip/hip_bf16.h>

typedef __attribute__((ext_vector_type(16))) __bf16 v16bf;
typedef __attribute__((ext_vector_type(8)))  float  v8f;

constexpr int  BATCH  = 4;
constexpr int  C      = 192;
constexpr int  C2     = 384;
constexpr int  HH     = 192;
constexpr int  WW     = 192;
constexpr long NPIX   = (long)HH * WW;   // 36864
constexpr int  HEADS  = 6;
constexpr int  CHUNKS = 8;
constexpr long CS     = NPIX / CHUNKS;   // 4608

__device__ __forceinline__ __bf16 to_bf16(float x)  { return (__bf16)x; }
__device__ __forceinline__ __bf16 to_bf16(__bf16 x) { return x; }

// Wave-uniform value -> SGPR so derived control flow stays scalar
// (keeps EXEC untouched around WMMA, moves index math to SALU).
__device__ __forceinline__ int uwave(int v) {
  return __builtin_amdgcn_readfirstlane(v);
}
__device__ __forceinline__ int wave_id() {
  return uwave((int)(blockIdx.x * (blockDim.x >> 5) + (threadIdx.x >> 5)));
}

__device__ __forceinline__ v8f wmma_bf16(v16bf a, v16bf b, v8f c) {
  // D = A*B + C, 16x16x32 bf16 -> f32 accumulate
  return __builtin_amdgcn_wmma_f32_16x16x32_bf16(
      /*neg_a=*/false, a, /*neg_b=*/false, b,
      /*c_mod=*/(short)0, c, /*reuse_a=*/false, /*reuse_b=*/false);
}

// ---------------------------------------------------------------------------
// 1x1 conv as GEMM:  Y[b][oc][n] = sum_k W[oc][k] * X[b][k][n]
// One wave computes a 64(oc) x 16(n) strip: B fragment loaded once per K-step
// and reused across 4 WMMA tiles. oc-quad is the fastest grid index so that
// concurrent waves share the same X columns (L2 absorbs M-direction rereads;
// X streams from HBM ~once).
// ---------------------------------------------------------------------------
template <typename TIN, typename TOUT>
__global__ void gemm1x1_kernel(const float* __restrict__ Wm,
                               const TIN*  __restrict__ X,
                               TOUT*       __restrict__ Y,
                               int OC, int K, int totalWaves) {
  const int gw = wave_id();
  if (gw >= totalWaves) return;                  // scalar branch
  const int lane = threadIdx.x & 31;
  const int l15  = lane & 15;

  const int nGroups = (int)(NPIX / 16);          // 2304 n-tiles
  const int ocQuads = OC >> 6;                   // 64-row M strips
  const int ocQ = gw % ocQuads;                  // SALU (gw is SGPR)
  const int r   = gw / ocQuads;
  const int nG  = r % nGroups;
  const int b   = r / nGroups;
  const int oc0 = ocQ << 6;
  const int n0  = nG << 4;

  const int koffA = (lane < 16) ? 0 : 16;
  const int koffB = (lane < 16) ? 0 : 16;
  const TIN* xb = X + (size_t)b * K * NPIX + n0 + l15;

  v8f acc[4] = {};
  for (int k0 = 0; k0 < K; k0 += 32) {
    const TIN* xcol = xb + (size_t)(k0 + koffB) * NPIX;
    if (k0 + 32 < K)
      __builtin_prefetch(xcol + (size_t)32 * NPIX, 0, 1);
    v16bf bf;
#pragma unroll
    for (int e = 0; e < 16; ++e)
      bf[e] = to_bf16(xcol[(size_t)e * NPIX]);
#pragma unroll
    for (int m4 = 0; m4 < 4; ++m4) {
      const float* wrow = Wm + (size_t)(oc0 + m4 * 16 + l15) * K + k0 + koffA;
      v16bf a;
#pragma unroll
      for (int e = 0; e < 16; ++e)
        a[e] = (__bf16)wrow[e];
      acc[m4] = wmma_bf16(a, bf, acc[m4]);
    }
  }
  const int moff = (lane < 16) ? 0 : 8;
#pragma unroll
  for (int m4 = 0; m4 < 4; ++m4)
#pragma unroll
    for (int rr = 0; rr < 8; ++rr) {
      size_t idx = ((size_t)b * OC + oc0 + m4 * 16 + rr + moff) * NPIX + n0 + l15;
      Y[idx] = (TOUT)acc[m4][rr];
    }
}

// ---------------------------------------------------------------------------
// Repack q weights [oc][ic][3][3] -> bf16 [tap][oc][ic] so the tap-major
// implicit GEMM gets contiguous A-fragment loads.
// ---------------------------------------------------------------------------
__global__ void repack_qw_kernel(const float* __restrict__ qw,
                                 __bf16* __restrict__ wt) {
  int idx = blockIdx.x * blockDim.x + threadIdx.x;     // < 9*192*192
  if (idx >= 9 * C * C) return;
  int tap = idx / (C * C);
  int rem = idx - tap * (C * C);
  int oc = rem / C;
  int ic = rem - oc * C;
  wt[idx] = (__bf16)qw[(size_t)(oc * C + ic) * 9 + tap];
}

// ---------------------------------------------------------------------------
// q 3x3 conv as 9 tap-GEMMs of K=192 accumulating into one set of WMMA
// accumulators. A 16-pixel tile never crosses an image row, so row validity
// per tap is a *scalar* branch (tile coords are SGPRs via readfirstlane;
// EXEC stays all-1s for WMMA) and column validity is a per-lane
// clamped-address + select (no divergence, no integer division).
// ---------------------------------------------------------------------------
__global__ void qconv3x3_kernel(const __bf16* __restrict__ Wt,  // [9][192][192]
                                const float*  __restrict__ X,   // [B][192][H][W]
                                __bf16*       __restrict__ Y,   // [B][192][NPIX]
                                int totalWaves) {
  const int gw = wave_id();
  if (gw >= totalWaves) return;                  // scalar branch
  const int lane = threadIdx.x & 31;
  const int l15  = lane & 15;

  const int nGroups = (int)(NPIX / 16);          // 2304
  const int ocQuads = C >> 6;                    // 3
  const int ocQ = gw % ocQuads;                  // SALU
  const int r   = gw / ocQuads;
  const int nG  = r % nGroups;
  const int b   = r / nGroups;
  const int oc0 = ocQ << 6;
  const int n0  = nG << 4;
  const int py  = n0 / WW;                       // SGPR: whole tile in one row
  const int px0 = n0 - py * WW;
  const int px  = px0 + l15;                     // per-lane column

  const int koffA = (lane < 16) ? 0 : 16;
  const int koffB = (lane < 16) ? 0 : 16;
  const float* xb = X + (size_t)b * C * NPIX;

  v8f acc[4] = {};
#pragma unroll
  for (int tap = 0; tap < 9; ++tap) {
    const int dh = tap / 3 - 1;                  // compile-time per unrolled iter
    const int dw = tap - (tap / 3) * 3 - 1;
    const int yy = py + dh;                      // SGPR
    if ((unsigned)yy >= (unsigned)HH) continue;  // scalar skip (s_cmp/s_cbranch)
    const int xx = px + dw;
    const bool colok = (unsigned)xx < (unsigned)WW;
    const int xxc = xx < 0 ? 0 : (xx > WW - 1 ? WW - 1 : xx);
    const float*  xrow = xb + (size_t)yy * WW + xxc;
    const __bf16* wtap = Wt + (size_t)tap * C * C;
    for (int k0 = 0; k0 < C; k0 += 32) {
      const float* xc = xrow + (size_t)(k0 + koffB) * NPIX;
      v16bf bf;
#pragma unroll
      for (int e = 0; e < 16; ++e) {
        float v = xc[(size_t)e * NPIX];
        bf[e] = (__bf16)(colok ? v : 0.f);
      }
#pragma unroll
      for (int m4 = 0; m4 < 4; ++m4) {
        const __bf16* wr = wtap + (size_t)(oc0 + m4 * 16 + l15) * C + k0 + koffA;
        v16bf a;
#pragma unroll
        for (int e = 0; e < 16; ++e)
          a[e] = wr[e];
        acc[m4] = wmma_bf16(a, bf, acc[m4]);
      }
    }
  }
  const int moff = (lane < 16) ? 0 : 8;
#pragma unroll
  for (int m4 = 0; m4 < 4; ++m4)
#pragma unroll
    for (int rr = 0; rr < 8; ++rr) {
      size_t idx = ((size_t)b * C + oc0 + m4 * 16 + rr + moff) * NPIX + n0 + l15;
      Y[idx] = (__bf16)acc[m4][rr];
    }
}

// ---------------------------------------------------------------------------
// Depthwise 3x3, groups = 384, SAME padding.
// ---------------------------------------------------------------------------
__global__ void dwconv3x3_kernel(const __bf16* __restrict__ X,  // [B][384][H][W]
                                 const float*  __restrict__ Wm, // [384][9]
                                 __bf16*       __restrict__ Y) {
  size_t idx = (size_t)blockIdx.x * blockDim.x + threadIdx.x;
  int px = (int)(idx % WW);
  size_t t = idx / WW;
  int py = (int)(t % HH);
  size_t p = t / HH;              // p = b*384 + ch
  int ch = (int)(p % C2);
  const __bf16* plane = X + p * NPIX;
  const float*  w     = Wm + (size_t)ch * 9;
  float s = 0.f;
#pragma unroll
  for (int kh = 0; kh < 3; ++kh) {
    int yy = py + kh - 1;
    if ((unsigned)yy >= (unsigned)HH) continue;
#pragma unroll
    for (int kw = 0; kw < 3; ++kw) {
      int xx = px + kw - 1;
      if ((unsigned)xx >= (unsigned)WW) continue;
      s += w[kh * 3 + kw] * (float)plane[(size_t)yy * WW + xx];
    }
  }
  Y[idx] = (__bf16)s;
}

// ---------------------------------------------------------------------------
// Per-(b,channel) sum of squares over spatial dim -> 1/max(norm, 1e-12).
// ---------------------------------------------------------------------------
__global__ void sumsq_inv_kernel(const __bf16* __restrict__ X, int chPerBatch,
                                 float* __restrict__ inv) {
  int p = blockIdx.x;                    // 0 .. B*192-1
  int b = p / C, c = p - b * C;
  const __bf16* plane = X + ((size_t)b * chPerBatch + c) * NPIX;
  float s = 0.f;
  for (int i = threadIdx.x; i < (int)NPIX; i += blockDim.x) {
    float v = (float)plane[i];
    s += v * v;
  }
  __shared__ float red[256];
  red[threadIdx.x] = s;
  __syncthreads();
  for (int off = 128; off > 0; off >>= 1) {
    if ((int)threadIdx.x < off) red[threadIdx.x] += red[threadIdx.x + off];
    __syncthreads();
  }
  if (threadIdx.x == 0) inv[p] = 1.f / fmaxf(sqrtf(red[0]), 1e-12f);
}

// ---------------------------------------------------------------------------
// attn_raw partials: per (b,h,chunk) block (4 waves -> 2x2 16x16 tiles),
// K-loop over the chunk's spatial range. Deterministic (no atomics).
// ---------------------------------------------------------------------------
__global__ void attn_qk_kernel(const __bf16* __restrict__ Q,   // [B][192][N]
                               const __bf16* __restrict__ KV,  // [B][384][N]; k = first 192
                               float* __restrict__ part) {     // [BH][CHUNKS][32][32]
  int blk = blockIdx.x;
  int ck  = blk % CHUNKS;
  int bh  = blk / CHUNKS;
  int b = bh / HEADS, h = bh - b * HEADS;
  int w = uwave(threadIdx.x >> 5);       // wave-uniform -> SGPR
  int lane = threadIdx.x & 31, l15 = lane & 15;
  int mh = w >> 1, dh = w & 1;
  const __bf16* qp = Q + ((size_t)b * C + h * 32 + mh * 16) * NPIX;
  const __bf16* kp = KV + ((size_t)b * C2 + h * 32 + dh * 16) * NPIX;
  const int koffA = (lane < 16) ? 0 : 16;
  const int koffB = (lane < 16) ? 0 : 16;

  v8f acc = {};
  const long kend = (long)(ck + 1) * CS;
  for (long k0 = (long)ck * CS; k0 < kend; k0 += 32) {
    const __bf16* qr = qp + (size_t)l15 * NPIX + k0 + koffA;
    const __bf16* kr = kp + (size_t)l15 * NPIX + k0 + koffB;
    __builtin_prefetch(qr + 32, 0, 1);
    __builtin_prefetch(kr + 32, 0, 1);
    v16bf a, bf;
#pragma unroll
    for (int e = 0; e < 16; ++e) a[e] = qr[e];
#pragma unroll
    for (int e = 0; e < 16; ++e) bf[e] = kr[e];
    acc = wmma_bf16(a, bf, acc);
  }
  const int moff = (lane < 16) ? 0 : 8;
#pragma unroll
  for (int rr = 0; rr < 8; ++rr) {
    int cr = mh * 16 + rr + moff;
    int dc = dh * 16 + l15;
    part[(((size_t)bh * CHUNKS + ck) * 32 + cr) * 32 + dc] = acc[rr];
  }
}

// ---------------------------------------------------------------------------
// Softmax row kernel: sums K-split partials, applies L2-norm scales and
// temperature, row softmax over 32 lanes (wave32 shuffles), emits bf16 attn.
// ---------------------------------------------------------------------------
__global__ void softmax_kernel(const float* __restrict__ part,
                               const float* __restrict__ invq,
                               const float* __restrict__ invk,
                               const float* __restrict__ temp,
                               __bf16* __restrict__ attn) {
  int gw = wave_id();
  if (gw >= BATCH * HEADS * 32) return;
  int lane = threadIdx.x & 31;
  int bh = gw >> 5, cr = gw & 31;
  int b = bh / HEADS, h = bh - b * HEADS;
  float raw = 0.f;
#pragma unroll
  for (int ck = 0; ck < CHUNKS; ++ck)
    raw += part[(((size_t)bh * CHUNKS + ck) * 32 + cr) * 32 + lane];
  float logit = raw * invq[b * C + h * 32 + cr] * invk[b * C + h * 32 + lane] * temp[h];
  float mx = logit;
  for (int off = 16; off > 0; off >>= 1) mx = fmaxf(mx, __shfl_xor(mx, off, 32));
  float e = __expf(logit - mx);
  float s = e;
  for (int off = 16; off > 0; off >>= 1) s += __shfl_xor(s, off, 32);
  attn[(size_t)bh * 1024 + cr * 32 + lane] = (__bf16)(e / s);
}

// ---------------------------------------------------------------------------
// out = attn @ v : per wave one 16(c) x 16(n) tile per WMMA; K = 32 (exact).
// A fragment (attn rows) loaded once, reused across 16 n-tiles.
// ---------------------------------------------------------------------------
__global__ void attnv_kernel(const __bf16* __restrict__ attn,  // [BH][32][32]
                             const __bf16* __restrict__ KV,    // v = planes 192..383
                             __bf16* __restrict__ Yout,        // [B][192][N]
                             int totalWaves) {
  const int gw = wave_id();
  if (gw >= totalWaves) return;
  const int lane = threadIdx.x & 31;
  const int l15  = lane & 15;
  const int NG = 16;
  const int groups = (int)(NPIX / (16 * NG));  // 144
  int ng  = gw % groups;                       // SALU
  int rem = gw / groups;
  int mh = rem & 1;
  int bh = rem >> 1;
  int b = bh / HEADS, h = bh - b * HEADS;
  const int koffA = (lane < 16) ? 0 : 16;
  const int koffB = (lane < 16) ? 0 : 16;

  v16bf a;
  const __bf16* ap = attn + (size_t)bh * 1024 + (size_t)(mh * 16 + l15) * 32 + koffA;
#pragma unroll
  for (int e = 0; e < 16; ++e) a[e] = ap[e];

  const __bf16* vp = KV + ((size_t)b * C2 + C + h * 32) * NPIX;
  __bf16* yb = Yout + ((size_t)b * C + h * 32 + mh * 16) * NPIX;
  const int moff = (lane < 16) ? 0 : 8;
  for (int t = 0; t < NG; ++t) {
    long n0 = ((long)ng * NG + t) * 16;
    v16bf bf;
    const __bf16* vr = vp + (size_t)koffB * NPIX + n0 + l15;
#pragma unroll
    for (int e = 0; e < 16; ++e) bf[e] = vr[(size_t)e * NPIX];
    v8f acc = {};
    acc = wmma_bf16(a, bf, acc);
#pragma unroll
    for (int rr = 0; rr < 8; ++rr)
      yb[(size_t)(rr + moff) * NPIX + n0 + l15] = (__bf16)acc[rr];
  }
}

// ---------------------------------------------------------------------------
extern "C" void kernel_launch(void* const* d_in, const int* in_sizes, int n_in,
                              void* d_out, int out_size, void* d_ws, size_t ws_size,
                              hipStream_t stream) {
  const float* x      = (const float*)d_in[0];
  const float* y      = (const float*)d_in[1];
  const float* q_w    = (const float*)d_in[2];
  const float* kv_w   = (const float*)d_in[3];
  const float* kvdw_w = (const float*)d_in[4];
  const float* proj_w = (const float*)d_in[5];
  const float* temp   = (const float*)d_in[6];
  float* out = (float*)d_out;
  (void)in_sizes; (void)n_in; (void)out_size; (void)ws_size;

  char* ws = (char*)d_ws;
  const size_t szKV = (size_t)BATCH * C2 * NPIX * sizeof(__bf16);  // 113.2 MB
  const size_t szQ  = (size_t)BATCH * C  * NPIX * sizeof(__bf16);  //  56.6 MB
  __bf16* kv_bf   = (__bf16*)(ws);
  __bf16* kvdw_bf = (__bf16*)(ws + szKV);
  __bf16* q_bf    = (__bf16*)(ws + 2 * szKV);
  char* p = ws + 2 * szKV + szQ;
  float* invq = (float*)p;  p += 4096;
  float* invk = (float*)p;  p += 4096;
  float* part = (float*)p;  p += (size_t)BATCH * HEADS * CHUNKS * 32 * 32 * sizeof(float);
  __bf16* attn_bf = (__bf16*)p;  p += (size_t)BATCH * HEADS * 32 * 32 * sizeof(__bf16);
  __bf16* wt_bf   = (__bf16*)p;  // 9*192*192 bf16 = 664 KB
  __bf16* attnout = kv_bf;       // kv no longer needed after depthwise -> reuse

  // 0. repack q weights to [tap][oc][ic] bf16
  repack_qw_kernel<<<(9 * C * C + 255) / 256, 256, 0, stream>>>(q_w, wt_bf);
  // 1. kv = 1x1 conv(y): [384x192] x [192xN] per batch
  {
    int totalWaves = BATCH * (int)(NPIX / 16) * (C2 / 64);
    gemm1x1_kernel<float, __bf16><<<(totalWaves + 7) / 8, 256, 0, stream>>>(
        kv_w, y, kv_bf, C2, C, totalWaves);
  }
  // 2. depthwise 3x3 (groups = 384)
  {
    size_t total = (size_t)BATCH * C2 * NPIX;
    dwconv3x3_kernel<<<(unsigned)(total / 256), 256, 0, stream>>>(kv_bf, kvdw_w, kvdw_bf);
  }
  // 3. q = 3x3 conv(x), tap-major implicit GEMM
  {
    int totalWaves = BATCH * (int)(NPIX / 16) * (C / 64);
    qconv3x3_kernel<<<(totalWaves + 7) / 8, 256, 0, stream>>>(wt_bf, x, q_bf, totalWaves);
  }
  // 4. L2 norms over spatial dim for q and k
  sumsq_inv_kernel<<<BATCH * C, 256, 0, stream>>>(q_bf, C, invq);
  sumsq_inv_kernel<<<BATCH * C, 256, 0, stream>>>(kvdw_bf, C2, invk);
  // 5. q k^T partials (K-split, deterministic)
  attn_qk_kernel<<<BATCH * HEADS * CHUNKS, 128, 0, stream>>>(q_bf, kvdw_bf, part);
  // 6. softmax (+ norm scales + temperature)
  softmax_kernel<<<(BATCH * HEADS * 32 + 7) / 8, 256, 0, stream>>>(
      part, invq, invk, temp, attn_bf);
  // 7. attn @ v
  {
    int totalWaves = BATCH * HEADS * 2 * (int)(NPIX / 256);
    attnv_kernel<<<(totalWaves + 7) / 8, 256, 0, stream>>>(attn_bf, kvdw_bf, attnout, totalWaves);
  }
  // 8. proj 1x1 conv -> f32 output
  {
    int totalWaves = BATCH * (int)(NPIX / 16) * (C / 64);
    gemm1x1_kernel<__bf16, float><<<(totalWaves + 7) / 8, 256, 0, stream>>>(
        proj_w, attnout, out, C, C, totalWaves);
  }
}
